// CAMILSelfAttention_51153060495546
// MI455X (gfx1250) — matrix operand order repaired and checked by reading, not compiled, not run on hardware
//
#include <hip/hip_runtime.h>
#include <hip/hip_bf16.h>

// Problem constants (B=1)
#define NTOK  8192
#define INDIM 1024
#define ATTD  512

typedef __bf16 bf16_t;
typedef __attribute__((ext_vector_type(16))) __bf16 v16bf;
typedef __attribute__((ext_vector_type(8)))  float  v8f;
typedef unsigned int u32;
typedef __attribute__((ext_vector_type(4))) unsigned int u32x4;
typedef __attribute__((ext_vector_type(8))) int i32x8;
typedef __attribute__((ext_vector_type(4))) int i32x4;

#if __has_builtin(__builtin_amdgcn_tensor_load_to_lds)
#define HAVE_TDM 1
#else
#define HAVE_TDM 0
#endif

// ---------------------------------------------------------------------------
// fp32 -> bf16 row-major copy
// ---------------------------------------------------------------------------
__global__ void cvt_bf16_kernel(const float* __restrict__ in,
                                bf16_t* __restrict__ out, int n) {
  int i = blockIdx.x * blockDim.x + threadIdx.x;
  if (i < n) out[i] = (bf16_t)in[i];
}

// fp32 [R x C] row-major -> bf16 column-major: out[c*R + r] = in[r*C + c]
__global__ void cvt_t_bf16_kernel(const float* __restrict__ in,
                                  bf16_t* __restrict__ out, int R, int C) {
  int i = blockIdx.x * blockDim.x + threadIdx.x;
  if (i >= R * C) return;
  int r = i / C, c = i % C;
  out[(size_t)c * R + r] = (bf16_t)in[i];
}

// ---------------------------------------------------------------------------
// BF16 WMMA GEMM (L2-resident operands): C[M,N] = A[M,K] * Bt[n][k].
// MODE 0 : full fp32 C (ld = N).
// MODE 1 : qk split -> cols < ATTD: fp32 Q (ld=ATTD); cols >= ATTD: bf16 K^T.
// ---------------------------------------------------------------------------
template<int MODE>
__global__ __launch_bounds__(128)
void gemm_wmma_bf16(const bf16_t* __restrict__ A,
                    const bf16_t* __restrict__ Bt,
                    float* __restrict__ C0,
                    bf16_t* __restrict__ C1,
                    int M, int N, int K)
{
  const int lane = threadIdx.x;        // 0..31 (wave32)
  const int wv   = threadIdx.y;        // 0..3
  const int m    = lane & 15;
  const int half = lane >> 4;

  const int rowBase = blockIdx.x * 64 + wv * 16;
  const int colBase = blockIdx.y * 64;

  const int kbA = 8 * half;    // A 16x32 bf16 layout: half-wave K split by 8
  const int kbB = 8 * half;    // B 32x16 bf16 layout: half-wave K split by 8

  v8f acc[4];
#pragma unroll
  for (int t = 0; t < 4; ++t) acc[t] = (v8f){0,0,0,0,0,0,0,0};

  const size_t arow = (size_t)(rowBase + m) * (size_t)K;
  const bf16_t* bcol[4];
#pragma unroll
  for (int t = 0; t < 4; ++t)
    bcol[t] = Bt + (size_t)(colBase + t * 16 + m) * (size_t)K;

  for (int k0 = 0; k0 < K; k0 += 32) {
    v16bf a;
    const bf16_t* ra = A + arow + k0;
#pragma unroll
    for (int p = 0; p < 4; ++p) {
      a[2*p]     = ra[kbA + 2*p];
      a[2*p + 1] = ra[kbA + 2*p + 1];
      a[8 + 2*p] = ra[16 + kbA + 2*p];
      a[9 + 2*p] = ra[16 + kbA + 2*p + 1];
    }
#pragma unroll
    for (int t = 0; t < 4; ++t) {
      const bf16_t* cb = bcol[t] + k0;
      v16bf b;
#pragma unroll
      for (int j = 0; j < 8; ++j) {
        b[2*j]     = cb[kbB + 2*j + ((j >> 2) << 3)];
        b[2*j + 1] = cb[kbB + 2*j + ((j >> 2) << 3) + 1];
      }
      acc[t] = __builtin_amdgcn_wmma_f32_16x16x32_bf16(
          false, a, false, b, (short)0, acc[t], false, false);
    }
  }

#pragma unroll
  for (int t = 0; t < 4; ++t) {
#pragma unroll
    for (int r = 0; r < 8; ++r) {
      int gr = rowBase + r + 8 * half;
      int gc = colBase + t * 16 + m;
      float v = acc[t][r];
      if (MODE == 0) {
        C0[(size_t)gr * (size_t)N + gc] = v;
      } else {
        if (gc < ATTD) C0[(size_t)gr * ATTD + gc] = v;                  // Q fp32
        else           C1[(size_t)(gc - ATTD) * NTOK + gr] = (bf16_t)v; // K^T bf16
      }
    }
  }
}

// ---------------------------------------------------------------------------
// TDM descriptor issue: 2D tile 64 rows x 64 fp32 from adj (stride 8192) into
// LDS with HW padding: 4 dwords every 64 dwords -> 68-float LDS row stride.
// (6-arg amdgpu-toolchain form: groups 2/3 + extra group unused for 2D tiles.)
// ---------------------------------------------------------------------------
#if HAVE_TDM
__device__ __forceinline__ void tdm_load_adj_tile(const float* gsrc, u32 lds_off) {
  unsigned long long ga = (unsigned long long)gsrc;
  u32x4 g0;
  g0.x = 1u;                               // count=1 (user descriptor)
  g0.y = lds_off;                          // LDS byte offset
  g0.z = (u32)ga;                          // global_addr[31:0]
  g0.w = (u32)(ga >> 32) | (2u << 30);     // global_addr[56:32] | type=2
  i32x8 g1;
  g1[0] = (2 << 16)          // data_size = 4B
        | (1 << 20)          // pad_enable
        | (5 << 22)          // pad_interval: 64 dwords
        | (3 << 25);         // pad_amount : 4 dwords
  g1[1] = (int)(0x2000u << 16);  // tensor_dim0 = 8192 (bits 79:48, low half)
  g1[2] = (int)(0x2000u << 16);  // tensor_dim0 hi=0 ; tensor_dim1 = 8192 low half
  g1[3] = (64 << 16);            // tensor_dim1 hi=0 ; tile_dim0 = 64
  g1[4] = 64;                    // tile_dim1 = 64 ; tile_dim2 = 0
  g1[5] = 8192;                  // tensor_dim0_stride low32 (elements)
  g1[6] = 0;                     // stride hi ; tensor_dim1_stride = 0 (2D)
  g1[7] = 0;
  i32x4 z4 = (i32x4){0, 0, 0, 0};
  i32x8 z8 = (i32x8){0, 0, 0, 0, 0, 0, 0, 0};
  __builtin_amdgcn_tensor_load_to_lds(g0, g1, z4, z4, z8, 0);
}
#endif

// ---------------------------------------------------------------------------
// adj-GEMM: S[8192,512] = adj[8192x8192] @ K[8192x512] (Kt col-major, bf16).
// One block = 64 rows x FULL 512 cols -> adj streamed from HBM exactly once.
// 16 waves: (wrow 0..3) x (wcol 0..3); wave = 16 rows x 128 cols (8 acc tiles).
// adj tile (64x64 fp32) double-buffered in LDS via TDM (or coop-copy fallback),
// converted fp32->bf16 (exact for {0,1}) while building A fragments.
// ---------------------------------------------------------------------------
#define LDSTRIDE 68   // 64 + 4 pad floats -> conflict-free ds_load_b128

__global__ __launch_bounds__(512)
void gemm_adj_wmma(const float* __restrict__ adj,
                   const bf16_t* __restrict__ Kt,
                   float* __restrict__ S)
{
  __shared__ float buf[2][64 * LDSTRIDE];

  const int lane = threadIdx.x;
  const int wv   = threadIdx.y;            // 0..15
  const int wrow = wv & 3;                 // row group (16 rows each)
  const int wcol = wv >> 2;                // col group (128 cols each)
  const int m    = lane & 15;
  const int half = lane >> 4;
  const int kbA  = 8 * half;
  const int kbB  = 8 * half;
  const int tid  = wv * 32 + lane;

  const int rowBase = blockIdx.x * 64;

  v8f acc[8];
#pragma unroll
  for (int t = 0; t < 8; ++t) acc[t] = (v8f){0,0,0,0,0,0,0,0};

  const bf16_t* bcol[8];
#pragma unroll
  for (int t = 0; t < 8; ++t)
    bcol[t] = Kt + (size_t)(wcol * 128 + t * 16 + m) * (size_t)NTOK;

  const int NSTAGE = NTOK / 64;            // 128 stages of K=64

#if HAVE_TDM
  if (wv == 0)
    tdm_load_adj_tile(adj + (size_t)rowBase * NTOK,
                      (u32)(unsigned long long)&buf[0][0]);
#endif

  for (int s = 0; s < NSTAGE; ++s) {
    const int cur = s & 1;
#if HAVE_TDM
    if (wv == 0) {
      if (s + 1 < NSTAGE) {
        tdm_load_adj_tile(adj + (size_t)rowBase * NTOK + (size_t)(s + 1) * 64,
                          (u32)(unsigned long long)&buf[cur ^ 1][0]);
        __builtin_amdgcn_s_wait_tensorcnt(1);   // stage s done (in-order)
      } else {
        __builtin_amdgcn_s_wait_tensorcnt(0);
      }
    }
#else
    // cooperative sync copy: 512 threads x 2 float4 = 64x64 fp32 tile
#pragma unroll
    for (int rep = 0; rep < 2; ++rep) {
      int c  = tid + rep * 512;
      int r  = c >> 4;
      int c4 = (c & 15) << 2;
      float4 v = *(const float4*)(adj + (size_t)(rowBase + r) * NTOK +
                                  (size_t)s * 64 + c4);
      *(float4*)&buf[cur][r * LDSTRIDE + c4] = v;
    }
#endif
    __syncthreads();

#pragma unroll
    for (int kk = 0; kk < 64; kk += 32) {
      // A fragment from LDS (fp32 -> bf16)
      const float* ap = &buf[cur][(wrow * 16 + m) * LDSTRIDE + kk];
      float4 f0 = *(const float4*)(ap + kbA);
      float4 f1 = *(const float4*)(ap + kbA + 4);
      float4 f2 = *(const float4*)(ap + 16 + kbA);
      float4 f3 = *(const float4*)(ap + 16 + kbA + 4);
      v16bf a;
      a[0]  = (bf16_t)f0.x; a[1]  = (bf16_t)f0.y; a[2]  = (bf16_t)f0.z; a[3]  = (bf16_t)f0.w;
      a[4]  = (bf16_t)f1.x; a[5]  = (bf16_t)f1.y; a[6]  = (bf16_t)f1.z; a[7]  = (bf16_t)f1.w;
      a[8]  = (bf16_t)f2.x; a[9]  = (bf16_t)f2.y; a[10] = (bf16_t)f2.z; a[11] = (bf16_t)f2.w;
      a[12] = (bf16_t)f3.x; a[13] = (bf16_t)f3.y; a[14] = (bf16_t)f3.z; a[15] = (bf16_t)f3.w;

      const int kglob = s * 64 + kk;
#pragma unroll
      for (int t = 0; t < 8; ++t) {
        const bf16_t* cb = bcol[t] + kglob;
        v16bf b;
#pragma unroll
        for (int j = 0; j < 8; ++j) {
          b[2*j]     = cb[kbB + 2*j + ((j >> 2) << 3)];
          b[2*j + 1] = cb[kbB + 2*j + ((j >> 2) << 3) + 1];
        }
        acc[t] = __builtin_amdgcn_wmma_f32_16x16x32_bf16(
            false, a, false, b, (short)0, acc[t], false, false);
      }
    }
    __syncthreads();
  }

#pragma unroll
  for (int t = 0; t < 8; ++t) {
#pragma unroll
    for (int r = 0; r < 8; ++r) {
      int gr = rowBase + wrow * 16 + r + 8 * half;
      int gc = wcol * 128 + t * 16 + m;
      S[(size_t)gr * ATTD + gc] = acc[t][r];
    }
  }
}

// ---------------------------------------------------------------------------
// w[n] = inv_scale * dot(Q[n,:], S[n,:])   (one wave per row)
// ---------------------------------------------------------------------------
__global__ void wdot_kernel(const float* __restrict__ Q,
                            const float* __restrict__ S,
                            float* __restrict__ w) {
  int row  = (blockIdx.x * blockDim.x + threadIdx.x) >> 5;
  int lane = threadIdx.x & 31;
  if (row >= NTOK) return;
  const float* q = Q + (size_t)row * ATTD;
  const float* s = S + (size_t)row * ATTD;
  float acc = 0.f;
  for (int i = lane; i < ATTD; i += 32) acc += q[i] * s[i];
#pragma unroll
  for (int off = 16; off > 0; off >>= 1) acc += __shfl_down(acc, off, 32);
  if (lane == 0) w[row] = acc * 0.04419417382415922f;  // 1/sqrt(512)
}

// ---------------------------------------------------------------------------
// softmax over the 8192-vector w (single block, 1024 threads = 32 waves)
// ---------------------------------------------------------------------------
__global__ void softmax_kernel(const float* __restrict__ w,
                               float* __restrict__ sm) {
  __shared__ float sred[32];
  const int tid  = threadIdx.x;
  const int lane = tid & 31;
  const int wv   = tid >> 5;

  float vals[8];
  float mx = -3.402823466e38f;
#pragma unroll
  for (int i = 0; i < 8; ++i) {
    vals[i] = w[tid + i * 1024];
    mx = fmaxf(mx, vals[i]);
  }
#pragma unroll
  for (int off = 16; off > 0; off >>= 1) mx = fmaxf(mx, __shfl_down(mx, off, 32));
  if (lane == 0) sred[wv] = mx;
  __syncthreads();
  float t = sred[lane];
#pragma unroll
  for (int off = 16; off > 0; off >>= 1) t = fmaxf(t, __shfl_down(t, off, 32));
  const float gmax = __shfl(t, 0, 32);
  __syncthreads();

  float ssum = 0.f;
#pragma unroll
  for (int i = 0; i < 8; ++i) {
    vals[i] = __expf(vals[i] - gmax);
    ssum += vals[i];
  }
#pragma unroll
  for (int off = 16; off > 0; off >>= 1) ssum += __shfl_down(ssum, off, 32);
  if (lane == 0) sred[wv] = ssum;
  __syncthreads();
  float u = sred[lane];
#pragma unroll
  for (int off = 16; off > 0; off >>= 1) u += __shfl_down(u, off, 32);
  const float inv = 1.f / __shfl(u, 0, 32);
#pragma unroll
  for (int i = 0; i < 8; ++i) sm[tid + i * 1024] = vals[i] * inv;
}

// ---------------------------------------------------------------------------
// L[n,d] = sm[n] * V[n,d]   (V already resident in d_out; scale in place)
// ---------------------------------------------------------------------------
__global__ void scale_kernel(float* __restrict__ out,
                             const float* __restrict__ sm) {
  int i = blockIdx.x * blockDim.x + threadIdx.x;     // over float4s
  const int n4 = NTOK * INDIM / 4;
  if (i >= n4) return;
  int row = i / (INDIM / 4);
  float s = sm[row];
  float4* o4 = (float4*)out;
  float4 v = o4[i];
  v.x *= s; v.y *= s; v.z *= s; v.w *= s;
  o4[i] = v;
}

// ---------------------------------------------------------------------------
extern "C" void kernel_launch(void* const* d_in, const int* in_sizes, int n_in,
                              void* d_out, int out_size, void* d_ws, size_t ws_size,
                              hipStream_t stream) {
  const float* X   = (const float*)d_in[0];   // (1, 8192, 1024)
  const float* adj = (const float*)d_in[1];   // (1, 8192, 8192)
  const float* Wqk = (const float*)d_in[2];   // (1024, 1024)
  const float* Wv  = (const float*)d_in[3];   // (1024, 1024)
  float* out = (float*)d_out;                 // (1, 8192, 1024)

  char* ws = (char*)d_ws;
  size_t off = 0;
  auto walloc = [&](size_t bytes) -> void* {
    void* p = ws + off;
    off += (bytes + 255) & ~(size_t)255;
    return p;
  };
  bf16_t* Xb   = (bf16_t*)walloc((size_t)NTOK * INDIM * 2);   // 16 MB
  bf16_t* Wqkt = (bf16_t*)walloc((size_t)INDIM * INDIM * 2);  //  2 MB (col-major)
  bf16_t* Wvt  = (bf16_t*)walloc((size_t)INDIM * INDIM * 2);  //  2 MB (col-major)
  float*  Qf   = (float*) walloc((size_t)NTOK * ATTD * 4);    // 16 MB
  bf16_t* Kt   = (bf16_t*)walloc((size_t)ATTD * NTOK * 2);    //  8 MB (col-major)
  float*  Sf   = (float*) walloc((size_t)NTOK * ATTD * 4);    // 16 MB
  float*  wbuf = (float*) walloc((size_t)NTOK * 4);
  float*  smb  = (float*) walloc((size_t)NTOK * 4);

  // 1) precision conversion (weights transposed for contiguous B-fragment loads)
  cvt_bf16_kernel<<<(NTOK * INDIM + 255) / 256, 256, 0, stream>>>(X, Xb, NTOK * INDIM);
  cvt_t_bf16_kernel<<<(INDIM * INDIM + 255) / 256, 256, 0, stream>>>(Wqk, Wqkt, INDIM, INDIM);
  cvt_t_bf16_kernel<<<(INDIM * INDIM + 255) / 256, 256, 0, stream>>>(Wv, Wvt, INDIM, INDIM);

  dim3 blk(32, 4);
  // 2) qk = X @ Wqk  -> Q (fp32) and K^T (bf16)
  gemm_wmma_bf16<1><<<dim3(NTOK / 64, INDIM / 64), blk, 0, stream>>>(
      Xb, Wqkt, Qf, Kt, NTOK, INDIM, INDIM);
  // 3) V = X @ Wv -> d_out (scaled in place later)
  gemm_wmma_bf16<0><<<dim3(NTOK / 64, INDIM / 64), blk, 0, stream>>>(
      Xb, Wvt, out, nullptr, NTOK, INDIM, INDIM);
  // 4) S = adj @ K  (adj streamed from HBM exactly once; TDM-staged in LDS)
  gemm_adj_wmma<<<dim3(NTOK / 64), dim3(32, 16), 0, stream>>>(adj, Kt, Sf);
  // 5) w[n] = inv_scale * q[n].s[n]
  wdot_kernel<<<NTOK / 8, 256, 0, stream>>>(Qf, Sf, wbuf);
  // 6) softmax over the N axis
  softmax_kernel<<<1, 1024, 0, stream>>>(wbuf, smb);
  // 7) L = sm[n] * V
  scale_kernel<<<(NTOK * INDIM / 4 + 255) / 256, 256, 0, stream>>>(out, smb);
}
